// PointNet2Encoder_67259187855923
// MI455X (gfx1250) — matrix operand order repaired and checked by reading, not compiled, not run on hardware
//
#include <hip/hip_runtime.h>
#include <hip/hip_bf16.h>

// ---------------------------------------------------------------------------
// PointNet++ encoder for MI455X (gfx1250).
// Live dataflow only (f1 / sa1-MLP are dead in the reference):
//   FPS(x,512) -> c1 ; FPS(c1,128) -> c2 ; ball_query(c2,c1,0.4,64)
//   grouped -> MLP 3->128->128->256 (WMMA f16, f32 acc, LDS-resident) -> max(64) -> f2
//   f2 -> MLP 256->256->512->1024 (WMMA) -> max(128) -> out (16,1024) f32
// All matrix math on v_wmma_f32_16x16x32_f16; 4 independent accumulators per
// wave (1 N-tile x 4 M-tiles) to share B-fragments and hide WMMA hazards.
// ---------------------------------------------------------------------------

typedef __attribute__((ext_vector_type(16))) _Float16 v16h;
typedef __attribute__((ext_vector_type(8)))  float    v8f;

union Frag16 { v16h v; uint4 q[2]; };
union Pack8  { uint4 q; _Float16 h[8]; };

constexpr int BB  = 16;    // batch
constexpr int N1  = 4096;  // input points
constexpr int G1  = 512;   // stage-1 centers
constexpr int G2  = 128;   // stage-2 centers
constexpr int S2  = 64;    // stage-2 group size

// ---------------------------------------------------------------------------
// 16x64 output block (1 N-tile x 4 M-tiles) of out = relu(W * actIn + bias).
// W      : [COUT][CINPAD] f16 row-major (BN-folded), CINPAD % 32 == 0
// actIn  : [P][CINPAD]    f16 point-major  (B-matrix: B[K=c][N=p])
// actOut : [P][COUT]      f16 point-major
// B-fragment shared across the 4 M-tiles; 4 independent accumulator chains.
// A/B use contiguous K = 16*half + 0..15 (same order in both); D rows 8*half + r (wave32).
// ---------------------------------------------------------------------------
template <int CINPAD, int COUT>
__device__ __forceinline__ void wmma_block_m4(
    const _Float16* __restrict__ W,
    const _Float16* __restrict__ actIn,
    const float*    __restrict__ bias,
    _Float16*       __restrict__ actOut,
    int n0, int m0, int lane)
{
  const int half = lane >> 4;
  const int mrow = m0 + (lane & 15);   // A-matrix row (M-tile 0) for this lane
  const int prow = n0 + (lane & 15);   // B/D point (N) for this lane
  v8f acc[4] = {v8f{}, v8f{}, v8f{}, v8f{}};
#pragma unroll
  for (int kb = 0; kb < CINPAD; kb += 32) {
    // B: elems 0..15 -> K = kb + 16*half + 0..15 (contiguous in [P][C])
    Frag16 b;
    const _Float16* bp = actIn + prow * CINPAD + kb + half * 16;
    b.q[0] = *(const uint4*)(bp);
    b.q[1] = *(const uint4*)(bp + 8);
#pragma unroll
    for (int t = 0; t < 4; ++t) {
      // A: elems 0..15 -> K = kb + 16*half + 0..15 (matches B)
      Frag16 a;
      const _Float16* wp = W + (mrow + t * 16) * CINPAD + kb + half * 16;
      a.q[0] = *(const uint4*)(wp);
      a.q[1] = *(const uint4*)(wp + 8);
      acc[t] = __builtin_amdgcn_wmma_f32_16x16x32_f16(false, a.v, false, b.v,
                                                      (short)0, acc[t],
                                                      false, false);
    }
  }
  // D: VGPR r -> M = m0 + t*16 + 8*half + r => 8 contiguous channels per store
#pragma unroll
  for (int t = 0; t < 4; ++t) {
    Pack8 o;
#pragma unroll
    for (int r = 0; r < 8; ++r) {
      const int M = m0 + t * 16 + half * 8 + r;
      float v = acc[t][r] + bias[M];
      o.h[r] = (_Float16)(v > 0.f ? v : 0.f);
    }
    *(uint4*)(actOut + prow * COUT + m0 + t * 16 + half * 8) = o.q;
  }
}

// ---------------------------------------------------------------------------
// Fold BatchNorm into conv weight/bias; emit f16 weights padded to Cinpad.
// ---------------------------------------------------------------------------
__global__ void fold_bn_kernel(const float* __restrict__ W,
                               const float* __restrict__ bias,
                               const float* __restrict__ gamma,
                               const float* __restrict__ beta,
                               const float* __restrict__ rm,
                               const float* __restrict__ rv,
                               _Float16* __restrict__ Wout,
                               float* __restrict__ bout,
                               int Cout, int Cin, int Cinpad)
{
  const int i = blockIdx.x * blockDim.x + threadIdx.x;
  if (i >= Cout * Cinpad) return;
  const int o = i / Cinpad, c = i % Cinpad;
  const float s = gamma[o] * rsqrtf(rv[o] + 1e-5f);
  Wout[i] = (c < Cin) ? (_Float16)(W[o * Cin + c] * s) : (_Float16)0.f;
  if (c == 0) bout[o] = bias[o] * s + beta[o] - s * rm[o];
}

// ---------------------------------------------------------------------------
// Iterative farthest point sampling: one workgroup per batch, dist in LDS.
// Exact reference semantics (init 1e10, first pick = point 0, first-index
// argmax tie-break).
// ---------------------------------------------------------------------------
__global__ __launch_bounds__(256) void fps_kernel(const float* __restrict__ pts,
                                                  float* __restrict__ centers,
                                                  int N, int G)
{
  __shared__ float dist[N1];
  __shared__ float rmax[256];
  __shared__ int   rarg[256];
  __shared__ int   cur;
  const int b = blockIdx.x, tid = threadIdx.x;
  for (int i = tid; i < N; i += 256) dist[i] = 1e10f;
  if (tid == 0) cur = 0;
  __syncthreads();
  const float* bp = pts + (size_t)b * N * 3;
  for (int it = 0; it < G; ++it) {
    const int c = cur;
    const float cx = bp[c * 3 + 0], cy = bp[c * 3 + 1], cz = bp[c * 3 + 2];
    if (tid == 0) {
      float* dst = centers + ((size_t)b * G + it) * 3;
      dst[0] = cx; dst[1] = cy; dst[2] = cz;
    }
    float best = -1.f; int bi = 0;
    for (int i = tid; i < N; i += 256) {
      const float dx = bp[i * 3 + 0] - cx;
      const float dy = bp[i * 3 + 1] - cy;
      const float dz = bp[i * 3 + 2] - cz;
      const float dd = fminf(dist[i], dx * dx + dy * dy + dz * dz);
      dist[i] = dd;
      if (dd > best) { best = dd; bi = i; }
    }
    rmax[tid] = best; rarg[tid] = bi;
    __syncthreads();
    for (int s = 128; s > 0; s >>= 1) {
      if (tid < s) {
        const float v = rmax[tid + s]; const int vi = rarg[tid + s];
        if (v > rmax[tid] || (v == rmax[tid] && vi < rarg[tid])) {
          rmax[tid] = v; rarg[tid] = vi;
        }
      }
      __syncthreads();
    }
    if (tid == 0) cur = rarg[0];
    __syncthreads();
  }
}

// ---------------------------------------------------------------------------
// Ball query: first-k-within-radius (ascending scan == sorted), pad w/ first.
// ---------------------------------------------------------------------------
__global__ void ball_query_kernel(const float* __restrict__ centers,
                                  const float* __restrict__ points,
                                  int* __restrict__ nidx,
                                  int G, int N, int K, float r2)
{
  const int i = blockIdx.x * blockDim.x + threadIdx.x;
  if (i >= BB * G) return;
  const int b = i / G;
  const float* cp = centers + (size_t)i * 3;
  const float cx = cp[0], cy = cp[1], cz = cp[2];
  const float* pp = points + (size_t)b * N * 3;
  int* out = nidx + (size_t)i * K;
  int cnt = 0, first = 0;
  for (int n = 0; n < N && cnt < K; ++n) {
    const float dx = pp[n * 3 + 0] - cx;
    const float dy = pp[n * 3 + 1] - cy;
    const float dz = pp[n * 3 + 2] - cz;
    if (dx * dx + dy * dy + dz * dz <= r2) {
      if (cnt == 0) first = n;
      out[cnt++] = n;
    }
  }
  for (; cnt < K; ++cnt) out[cnt] = first;
}

// ---------------------------------------------------------------------------
// Fused stage-2: gather group -> 3-layer WMMA MLP entirely in LDS -> max(64).
// One 8-wave workgroup per (batch, group). LDS: 16KB + 32KB ping-pong.
// Wave w handles N-tile (w&3), M-tiles 4*(w>>2)..4*(w>>2)+3.
// ---------------------------------------------------------------------------
__global__ __launch_bounds__(256) void sa2_mlp_kernel(
    const float* __restrict__ c1, const float* __restrict__ c2,
    const int* __restrict__ nidx,
    const _Float16* __restrict__ W0, const float* __restrict__ b0,
    const _Float16* __restrict__ W1, const float* __restrict__ b1,
    const _Float16* __restrict__ W2, const float* __restrict__ b2,
    _Float16* __restrict__ f2act)
{
  __shared__ __align__(32) _Float16 actA[S2 * 128];  // act0(stride 32) / L1 out
  __shared__ __align__(32) _Float16 actB[S2 * 256];  // L0 out / L2 out
  const int b = blockIdx.x / G2;
  const int g = blockIdx.x % G2;
  const int tid = threadIdx.x;

  // Build grouped relative coords: [64 pts][32 padded ch] f16 into actA.
  if (tid < S2) {
    const int idx = nidx[((size_t)(b * G2 + g)) * S2 + tid];
    const float cx = c2[(size_t)(b * G2 + g) * 3 + 0];
    const float cy = c2[(size_t)(b * G2 + g) * 3 + 1];
    const float cz = c2[(size_t)(b * G2 + g) * 3 + 2];
    _Float16* row = actA + tid * 32;
    row[0] = (_Float16)(c1[((size_t)b * G1 + idx) * 3 + 0] - cx);
    row[1] = (_Float16)(c1[((size_t)b * G1 + idx) * 3 + 1] - cy);
    row[2] = (_Float16)(c1[((size_t)b * G1 + idx) * 3 + 2] - cz);
    for (int c = 3; c < 32; ++c) row[c] = (_Float16)0.f;
  }
  __syncthreads();

  const int wave = tid >> 5, lane = tid & 31;
  // L0: 32 -> 128 (actA -> actB): 8 wave-jobs (4 N-tiles x 2 M-groups)
  wmma_block_m4<32, 128>(W0, actA, b0, actB,
                         (wave & 3) << 4, (wave >> 2) << 6, lane);
  __syncthreads();
  // L1: 128 -> 128 (actB -> actA)
  wmma_block_m4<128, 128>(W1, actB, b1, actA,
                          (wave & 3) << 4, (wave >> 2) << 6, lane);
  __syncthreads();
  // L2: 128 -> 256 (actA -> actB): 16 wave-jobs, 2 per wave
#pragma unroll
  for (int j = 0; j < 2; ++j) {
    const int job = wave + j * 8;
    wmma_block_m4<128, 256>(W2, actA, b2, actB,
                            (job & 3) << 4, (job >> 2) << 6, lane);
  }
  __syncthreads();

  // Max-pool over the 64 group points: one channel per thread.
  float m = (float)actB[tid];
  for (int p = 1; p < S2; ++p) m = fmaxf(m, (float)actB[p * 256 + tid]);
  f2act[((size_t)(b * G2 + g)) * 256 + tid] = (_Float16)m;
}

// ---------------------------------------------------------------------------
// Generic WMMA layer over global f16 activations: one 16x64 block per wave.
// ---------------------------------------------------------------------------
template <int CIN, int COUT>
__global__ __launch_bounds__(256) void mlp_layer_kernel(
    const _Float16* __restrict__ actIn, _Float16* __restrict__ actOut,
    const _Float16* __restrict__ W, const float* __restrict__ bias, int P)
{
  const int job  = (blockIdx.x * 256 + threadIdx.x) >> 5;
  const int lane = threadIdx.x & 31;
  const int ntiles_n = P >> 4;
  if (job >= (COUT >> 6) * ntiles_n) return;  // wave-uniform exit (EXEC all-1)
  const int n0 = (job % ntiles_n) << 4;
  const int m0 = (job / ntiles_n) << 6;
  wmma_block_m4<CIN, COUT>(W, actIn, bias, actOut, n0, m0, lane);
}

// ---------------------------------------------------------------------------
// Final max over the 128 groups per batch -> (16,1024) f32.
// ---------------------------------------------------------------------------
__global__ void final_max_kernel(const _Float16* __restrict__ act,
                                 float* __restrict__ out)
{
  const int i = blockIdx.x * blockDim.x + threadIdx.x;
  if (i >= BB * 1024) return;
  const int b = i >> 10, o = i & 1023;
  const _Float16* p = act + ((size_t)b * G2) * 1024 + o;
  float m = (float)p[0];
  for (int g = 1; g < G2; ++g) m = fmaxf(m, (float)p[(size_t)g * 1024]);
  out[i] = m;
}

// ---------------------------------------------------------------------------
extern "C" void kernel_launch(void* const* d_in, const int* in_sizes, int n_in,
                              void* d_out, int out_size, void* d_ws, size_t ws_size,
                              hipStream_t stream) {
  (void)in_sizes; (void)n_in; (void)out_size; (void)ws_size;
  // Input layout (setup_inputs insertion order):
  //   d_in[0] = x (16,4096,3) f32
  //   then 9 layers x {w,b,gamma,beta,rm,rv}: sa1 = layers 0..2 (dead),
  //   sa2 = layers 3..5, sa3 = layers 6..8.
  const float* x = (const float*)d_in[0];
  auto LP = [&](int j, int k) { return (const float*)d_in[1 + 6 * j + k]; };

  char* ws = (char*)d_ws;
  size_t off = 0;
  auto alloc = [&](size_t bytes) -> void* {
    off = (off + 255) & ~(size_t)255;
    void* p = (void*)(ws + off);
    off += bytes;
    return p;
  };

  float*    c1    = (float*)   alloc((size_t)BB * G1 * 3 * 4);
  float*    c2    = (float*)   alloc((size_t)BB * G2 * 3 * 4);
  int*      nidx2 = (int*)     alloc((size_t)BB * G2 * S2 * 4);
  _Float16* f2act = (_Float16*)alloc((size_t)BB * G2 * 256 * 2);
  _Float16* g3a   = (_Float16*)alloc((size_t)BB * G2 * 1024 * 2);
  _Float16* g3b   = (_Float16*)alloc((size_t)BB * G2 * 512 * 2);
  _Float16* W20   = (_Float16*)alloc(128 * 32  * 2);
  _Float16* W21   = (_Float16*)alloc(128 * 128 * 2);
  _Float16* W22   = (_Float16*)alloc(256 * 128 * 2);
  _Float16* W30   = (_Float16*)alloc(256 * 256 * 2);
  _Float16* W31   = (_Float16*)alloc(512 * 256 * 2);
  _Float16* W32   = (_Float16*)alloc(1024 * 512 * 2);
  float*    b20   = (float*)   alloc(128 * 4);
  float*    b21   = (float*)   alloc(128 * 4);
  float*    b22   = (float*)   alloc(256 * 4);
  float*    b30   = (float*)   alloc(256 * 4);
  float*    b31   = (float*)   alloc(512 * 4);
  float*    b32   = (float*)   alloc(1024 * 4);

  // --- Fold BN into weights (sa2 = layers 3..5, sa3 = layers 6..8) ---------
  struct FD { int j, Cout, Cin, Cinpad; _Float16* W; float* b; };
  const FD folds[6] = {
    {3, 128,  3,   32,  W20, b20}, {4, 128, 128, 128, W21, b21},
    {5, 256, 128, 128,  W22, b22}, {6, 256, 256, 256, W30, b30},
    {7, 512, 256, 256,  W31, b31}, {8, 1024, 512, 512, W32, b32},
  };
  for (int f = 0; f < 6; ++f) {
    const FD& d = folds[f];
    const int n = d.Cout * d.Cinpad;
    fold_bn_kernel<<<(n + 255) / 256, 256, 0, stream>>>(
        LP(d.j, 0), LP(d.j, 1), LP(d.j, 2), LP(d.j, 3), LP(d.j, 4), LP(d.j, 5),
        d.W, d.b, d.Cout, d.Cin, d.Cinpad);
  }

  // --- Geometry: two FPS passes + ball query -------------------------------
  fps_kernel<<<BB, 256, 0, stream>>>(x,  c1, N1, G1);
  fps_kernel<<<BB, 256, 0, stream>>>(c1, c2, G1, G2);
  ball_query_kernel<<<(BB * G2 + 255) / 256, 256, 0, stream>>>(
      c2, c1, nidx2, G2, G1, S2, 0.4f * 0.4f);

  // --- Stage-2 fused gather + 3-layer WMMA MLP + max(64) -------------------
  sa2_mlp_kernel<<<BB * G2, 256, 0, stream>>>(
      c1, c2, nidx2, W20, b20, W21, b21, W22, b22, f2act);

  // --- sa3 MLP over all 2048 points: 256->256->512->1024 -------------------
  const int P = BB * G2;  // 2048
  {
    int jobs = (256 >> 6) * (P >> 4);    // 512 wave-jobs -> 64 blocks
    mlp_layer_kernel<256, 256><<<(jobs + 7) / 8, 256, 0, stream>>>(f2act, g3a, W30, b30, P);
    jobs = (512 >> 6) * (P >> 4);        // 1024 -> 128 blocks
    mlp_layer_kernel<256, 512><<<(jobs + 7) / 8, 256, 0, stream>>>(g3a, g3b, W31, b31, P);
    jobs = (1024 >> 6) * (P >> 4);       // 2048 -> 256 blocks
    mlp_layer_kernel<512, 1024><<<(jobs + 7) / 8, 256, 0, stream>>>(g3b, g3a, W32, b32, P);
  }

  // --- Final max over groups -> (16,1024) f32 ------------------------------
  final_max_kernel<<<(BB * 1024 + 255) / 256, 256, 0, stream>>>(g3a, (float*)d_out);
}
